// SNNClassifier_51599737094640
// MI455X (gfx1250) — matrix operand order repaired and checked by reading, not compiled, not run on hardware
//
#include <hip/hip_runtime.h>

// SNN time-to-first-spike classifier, reduced to: exp -> GEMM1(f16 WMMA) ->
// rational epilogue -> GEMM2(f16 WMMA) -> log epilogue.  The argsort/scatter
// pairs in the reference cancel exactly, so no sorting is needed.
//
// GEMM1 stages its block tiles in LDS with the gfx1250 Tensor Data Mover
// (tensor_load_to_lds, TENSORcnt), double-buffered (explicit ping-pong so all
// LDS offsets are immediates), with TDM row padding to an 80B pitch to
// de-conflict the ds_load_b128 fragment reads.

typedef __attribute__((ext_vector_type(16))) _Float16     v16h;
typedef __attribute__((ext_vector_type(8)))  _Float16     v8h;
typedef __attribute__((ext_vector_type(8)))  float        v8f;
typedef __attribute__((ext_vector_type(4)))  unsigned int u32x4;
typedef __attribute__((ext_vector_type(8)))  unsigned int u32x8;

constexpr int BATCH   = 4096;
constexpr int INPUT   = 1024;
constexpr int HIDDEN  = 4096;
constexpr int CLASSES = 128;

// LDS tile geometry for GEMM1 (K-step = 32 f16 = 64B rows, padded to 80B)
constexpr unsigned LDS_PITCH = 80;                 // 64B row + 16B TDM pad
constexpr unsigned A_TILE_B  = 256 * LDS_PITCH;    // 20480 B
constexpr unsigned B_TILE_B  = 128 * LDS_PITCH;    // 10240 B
constexpr unsigned AS0 = 0;
constexpr unsigned AS1 = A_TILE_B;
constexpr unsigned BS0 = 2 * A_TILE_B;
constexpr unsigned BS1 = 2 * A_TILE_B + B_TILE_B;
constexpr unsigned SMEM_B = 2 * (A_TILE_B + B_TILE_B);   // 61440 B

// ---------------------------------------------------------------------------
// Prologue: h = (f16)exp(-x); W1,W2 -> f16 copies in workspace.
// ---------------------------------------------------------------------------
__global__ void snn_prep(const float* __restrict__ x,
                         const float* __restrict__ W1,
                         const float* __restrict__ W2,
                         _Float16* __restrict__ H16,
                         _Float16* __restrict__ W1h,
                         _Float16* __restrict__ W2h) {
    const int N1 = BATCH * INPUT;
    const int N2 = HIDDEN * INPUT;
    const int N3 = CLASSES * HIDDEN;
    int idx = blockIdx.x * blockDim.x + threadIdx.x;
    if (idx < N1) {
        H16[idx] = (_Float16)__expf(-x[idx]);
    } else if (idx < N1 + N2) {
        int i = idx - N1;
        W1h[i] = (_Float16)W1[i];
    } else if (idx < N1 + N2 + N3) {
        int i = idx - N1 - N2;
        W2h[i] = (_Float16)W2[i];
    }
}

// ---------------------------------------------------------------------------
// Tensor Data Mover: DMA one [tile_rows x 32] f16 tile (row pitch 1024 elems)
// from global into LDS at lds_byte_addr, padding each 64B row to 80B.
// 2D descriptor -> 2 SGPR groups.  Issued once per wave (EXEC ignored).
// ---------------------------------------------------------------------------
__device__ inline void tdm_load_tile_f16(const void* gptr,
                                         unsigned lds_byte_addr,
                                         unsigned tile_rows) {
    const unsigned long long ga = (unsigned long long)(size_t)gptr;
    u32x4 g0;
    g0[0] = 1u;                                      // count=1 (valid), user mode
    g0[1] = lds_byte_addr;                           // D#.lds_addr
    g0[2] = (unsigned)ga;                            // D#.global_addr[31:0]
    g0[3] = ((unsigned)(ga >> 32) & 0x01FFFFFFu)     // D#.global_addr[56:32]
            | (2u << 30);                            // D#.type = 2
    u32x8 g1;
    g1[0] = (1u << 16)       // data_size = 2 bytes
          | (1u << 20)       // pad_enable
          | (3u << 22)       // pad_interval = 16 DWORDs (64B)
          | (3u << 25);      // pad_amount   = 4 DWORDs (16B) -> 80B pitch
    g1[1] = ((unsigned)INPUT & 0xFFFFu) << 16;       // tensor_dim0 lo16 (=1024)
    g1[2] = ((unsigned)HIDDEN & 0xFFFFu) << 16;      // dim0 hi=0 | tensor_dim1 lo16 (=4096)
    g1[3] = (32u << 16);                             // dim1 hi=0 | tile_dim0 = 32
    g1[4] = tile_rows & 0xFFFFu;                     // tile_dim1 | tile_dim2=0
    g1[5] = (unsigned)INPUT;                         // tensor_dim0_stride lo32 (=1024)
    g1[6] = 0u;                                      // stride hi | dim1_stride lo
    g1[7] = 0u;
    asm volatile("tensor_load_to_lds %0, %1" :: "s"(g0), "s"(g1) : "memory");
}

// ---------------------------------------------------------------------------
// WMMA fragment helpers (wave32; A/B use contiguous K order, same for both, not the ISA table order).
// A (16x32 f16): lane&15 = row M; lanes 0-15 carry K 0-15,
//                lanes 16-31 carry K 16-31.
// B (32x16 f16): lane&15 = col N; lanes 0-15 carry K 0-15,
//                lanes 16-31 carry K 16-31 (rows of W are columns of B).
// C/D (16x16 f32): VGPR r -> M = r + (lane>=16 ? 8 : 0), N = lane&15.
// ---------------------------------------------------------------------------
__device__ inline v16h lds_frag_a(const char* smem, unsigned base, int row, int lane) {
    const int m   = lane & 15;
    const int klo = (lane & 16) ? 16 : 0;            // halfs
    const char* p = smem + base + (unsigned)(row + m) * LDS_PITCH + klo * 2;
    v8h lo = *(const v8h*)p;                         // K = klo .. klo+7
    v8h hi = *(const v8h*)(p + 16);                  // K = klo+8 .. klo+15
    v16h r;
#pragma unroll
    for (int i = 0; i < 8; ++i) { r[i] = lo[i]; r[i + 8] = hi[i]; }
    return r;
}

__device__ inline v16h lds_frag_b(const char* smem, unsigned base, int col, int lane) {
    const int n    = lane & 15;
    const int koff = (lane & 16) ? 32 : 0;           // bytes
    return *(const v16h*)(smem + base + (unsigned)(col + n) * LDS_PITCH + koff);
}

__device__ inline v16h load_frag_a(const _Float16* __restrict__ base, int ld,
                                   int row, int k0, int lane) {
    const int m   = lane & 15;
    const int klo = (lane & 16) ? 16 : 0;
    const _Float16* p = base + (size_t)(row + m) * ld + k0 + klo;
    v8h lo = *(const v8h*)p;
    v8h hi = *(const v8h*)(p + 8);
    v16h r;
#pragma unroll
    for (int i = 0; i < 8; ++i) { r[i] = lo[i]; r[i + 8] = hi[i]; }
    return r;
}

__device__ inline v16h load_frag_b(const _Float16* __restrict__ base, int ld,
                                   int col, int k0, int lane) {
    const int n    = lane & 15;
    const int koff = (lane & 16) ? 16 : 0;
    return *(const v16h*)(base + (size_t)(col + n) * ld + k0 + koff);
}

__device__ inline v8f wmma_f16(v16h a, v16h b, v8f c) {
    return __builtin_amdgcn_wmma_f32_16x16x32_f16(
        false, a, false, b, (short)0, c, false, false);
}

// One K-step of the wave-level 64x64 tile: 8 LDS fragment reads, 16 WMMAs.
__device__ inline void gemm1_step(const char* smem, unsigned abase, unsigned bbase,
                                  int waveM, int waveN, int lane, v8f acc[4][4]) {
    v16h a[4], b[4];
#pragma unroll
    for (int i = 0; i < 4; ++i)
        a[i] = lds_frag_a(smem, abase, waveM * 64 + 16 * i, lane);
#pragma unroll
    for (int j = 0; j < 4; ++j)
        b[j] = lds_frag_b(smem, bbase, waveN * 64 + 16 * j, lane);
#pragma unroll
    for (int i = 0; i < 4; ++i)
#pragma unroll
        for (int j = 0; j < 4; ++j)
            acc[i][j] = wmma_f16(a[i], b[j], acc[i][j]);
}

// ---------------------------------------------------------------------------
// GEMM1: p1 = H16 @ W1h^T  (4096x1024 @ 1024x4096), fused LIF epilogue
//        a = p1>1 ? (p1-1)/p1 : 0, stored f16.
// Block = 8 waves, block tile 256(M)x128(N), wave tile 64x64 (16 accums).
// Tiles staged via TDM into double-buffered LDS; wave 0 DMAs the A tile,
// wave 1 the B tile.  Explicit ping-pong keeps every LDS offset immediate.
// ---------------------------------------------------------------------------
__global__ __launch_bounds__(256)
void snn_gemm1(const _Float16* __restrict__ H16,
               const _Float16* __restrict__ W1h,
               _Float16* __restrict__ A2h) {
    __shared__ char smem[SMEM_B];
    const int lane  = threadIdx.x & 31;
    const int wave  = threadIdx.x >> 5;
    const int waveM = wave >> 1;   // 0..3
    const int waveN = wave & 1;    // 0..1
    const int row0  = blockIdx.y * 256;
    const int col0  = blockIdx.x * 128;

    // LDS aperture keeps the wave-relative byte offset in addr[31:0].
    const unsigned lds0 = (unsigned)(size_t)(void*)smem;
    const _Float16* Abase = H16 + (size_t)row0 * INPUT;
    const _Float16* Bbase = W1h + (size_t)col0 * INPUT;

    if (wave == 0) tdm_load_tile_f16(Abase, lds0 + AS0, 256);   // tile k=0
    else if (wave == 1) tdm_load_tile_f16(Bbase, lds0 + BS0, 128);

    v8f acc[4][4] = {};
    for (int k0 = 0; k0 < INPUT; k0 += 64) {
        // ---- sub-step 0: compute buf0, DMA k0+32 into buf1 ----
        __builtin_amdgcn_s_wait_tensorcnt(0);   // SALU: no-op for non-DMA waves
        __syncthreads();                        // tile visible; old reads retired
        if (wave == 0)      tdm_load_tile_f16(Abase + k0 + 32, lds0 + AS1, 256);
        else if (wave == 1) tdm_load_tile_f16(Bbase + k0 + 32, lds0 + BS1, 128);
        gemm1_step(smem, AS0, BS0, waveM, waveN, lane, acc);

        // ---- sub-step 1: compute buf1, DMA k0+64 into buf0 ----
        __builtin_amdgcn_s_wait_tensorcnt(0);
        __syncthreads();
        if (k0 + 64 < INPUT) {
            if (wave == 0)      tdm_load_tile_f16(Abase + k0 + 64, lds0 + AS0, 256);
            else if (wave == 1) tdm_load_tile_f16(Bbase + k0 + 64, lds0 + BS0, 128);
        }
        gemm1_step(smem, AS1, BS1, waveM, waveN, lane, acc);
    }

    // Epilogue: a = fired ? (p-1)/p : 0   (== exp(-t1))
    const int n_in_tile = lane & 15;
    const int m_half    = (lane & 16) ? 8 : 0;
#pragma unroll
    for (int i = 0; i < 4; ++i) {
#pragma unroll
        for (int j = 0; j < 4; ++j) {
            const int nn = col0 + waveN * 64 + 16 * j + n_in_tile;
            const int mb = row0 + waveM * 64 + 16 * i + m_half;
#pragma unroll
            for (int r = 0; r < 8; ++r) {
                float p   = acc[i][j][r];
                float val = (p > 1.0f) ? (p - 1.0f) / p : 0.0f;
                A2h[(size_t)(mb + r) * HIDDEN + nn] = (_Float16)val;
            }
        }
    }
}

// ---------------------------------------------------------------------------
// GEMM2: p2 = A2h @ W2h^T  (4096x4096 @ 4096x128), fused epilogue
//        out = p2>1 ? log(p2/(p2-1)) : inf, stored f32.
// Block = 8 waves, block tile 32(M)x128(N), wave tile 16x32 (2 accums).
// Small (4.3 GFLOP) and W2 is L2-hot: direct global fragment loads.
// ---------------------------------------------------------------------------
__global__ __launch_bounds__(256)
void snn_gemm2(const _Float16* __restrict__ A2h,
               const _Float16* __restrict__ W2h,
               float* __restrict__ out) {
    const int lane  = threadIdx.x & 31;
    const int wave  = threadIdx.x >> 5;
    const int waveM = wave >> 2;   // 0..1
    const int waveN = wave & 3;    // 0..3
    const int row0  = blockIdx.x * 32 + waveM * 16;
    const int col0  = waveN * 32;

    v8f acc0 = {}, acc1 = {};

    for (int k0 = 0; k0 < HIDDEN; k0 += 32) {
        v16h a  = load_frag_a(A2h, HIDDEN, row0, k0, lane);
        v16h b0 = load_frag_b(W2h, HIDDEN, col0,      k0, lane);
        v16h b1 = load_frag_b(W2h, HIDDEN, col0 + 16, k0, lane);
        acc0 = wmma_f16(a, b0, acc0);
        acc1 = wmma_f16(a, b1, acc1);
    }

    const int n_in_tile = lane & 15;
    const int mb        = row0 + ((lane & 16) ? 8 : 0);
#pragma unroll
    for (int t = 0; t < 2; ++t) {
        const v8f& acc = t ? acc1 : acc0;
        const int  nn  = col0 + 16 * t + n_in_tile;
#pragma unroll
        for (int r = 0; r < 8; ++r) {
            float p   = acc[r];
            float val = (p > 1.0f) ? __logf(p / (p - 1.0f))
                                   : __builtin_inff();
            out[(size_t)(mb + r) * CLASSES + nn] = val;
        }
    }
}

// ---------------------------------------------------------------------------
extern "C" void kernel_launch(void* const* d_in, const int* in_sizes, int n_in,
                              void* d_out, int out_size, void* d_ws, size_t ws_size,
                              hipStream_t stream) {
    (void)in_sizes; (void)n_in; (void)out_size; (void)ws_size;

    const float* x  = (const float*)d_in[0];   // (4096, 1024)
    const float* W1 = (const float*)d_in[1];   // (4096, 1024)
    const float* W2 = (const float*)d_in[2];   // (128, 4096)
    float* out = (float*)d_out;                // (4096, 128)

    // Workspace layout (f16): H16 | W1h | W2h | A2h  -> 49 MB total
    _Float16* H16 = (_Float16*)d_ws;
    _Float16* W1h = H16 + (size_t)BATCH * INPUT;
    _Float16* W2h = W1h + (size_t)HIDDEN * INPUT;
    _Float16* A2h = W2h + (size_t)CLASSES * HIDDEN;

    const int Ntot = BATCH * INPUT + HIDDEN * INPUT + CLASSES * HIDDEN;
    snn_prep<<<(Ntot + 255) / 256, 256, 0, stream>>>(x, W1, W2, H16, W1h, W2h);

    snn_gemm1<<<dim3(HIDDEN / 128, BATCH / 256), 256, 0, stream>>>(H16, W1h, A2h);

    snn_gemm2<<<dim3(BATCH / 32), 256, 0, stream>>>(A2h, W2h, out);
}
